// SingleHeadSelfAttention_77025943486840
// MI455X (gfx1250) — matrix operand order repaired and checked by reading, not compiled, not run on hardware
//
#include <hip/hip_runtime.h>

// MI455X / gfx1250: wave32, WMMA f16 -> f32 accum, 16x16x32 shape.
typedef __attribute__((ext_vector_type(16))) _Float16 v16h;
typedef __attribute__((ext_vector_type(8)))  float    v8f;
typedef __attribute__((ext_vector_type(4)))  int      v4i;

#define WMMA_F32_F16(a, b, c) \
  __builtin_amdgcn_wmma_f32_16x16x32_f16(false, (a), false, (b), (short)0, (c), false, false)

#define AS1 __attribute__((address_space(1)))
#define AS3 __attribute__((address_space(3)))

constexpr int Bb = 8, Tt = 2048, Cc = 768, Dd = 64;

// ---- CDNA5 async global->LDS copy (ASYNCcnt-tracked), with sync fallback ----
#if __has_builtin(__builtin_amdgcn_global_load_async_to_lds_b128)
__device__ __forceinline__ void async_cp32B(const _Float16* g, _Float16* l) {
  // two b128 copies; imm offset applies to both global and LDS addresses
  __builtin_amdgcn_global_load_async_to_lds_b128((AS1 v4i*)g, (AS3 v4i*)l, 0, 0);
  __builtin_amdgcn_global_load_async_to_lds_b128((AS1 v4i*)g, (AS3 v4i*)l, 16, 0);
}
#if __has_builtin(__builtin_amdgcn_s_wait_asynccnt)
#define WAIT_ASYNC(n) __builtin_amdgcn_s_wait_asynccnt(n)
#else
#define WAIT_ASYNC(n) asm volatile("s_wait_asynccnt %0" ::"i"(n) : "memory")
#endif
#else
__device__ __forceinline__ void async_cp32B(const _Float16* g, _Float16* l) {
  *(uint4*)l = *(const uint4*)g;
  *(uint4*)(l + 8) = *(const uint4*)(g + 8);
}
#define WAIT_ASYNC(n) ((void)0)
#endif

// ---------------------------------------------------------------------------
// Kernel 1: y[row, n] = sum_c x[row, c] * W[n, c], emitted as f16 (scale folded)
// block = 256 threads (8 waves), tile = 128 rows x 64 cols, K-step = 32.
// ---------------------------------------------------------------------------
__global__ __launch_bounds__(256)
void qkv_proj_kernel(const float* __restrict__ x, const float* __restrict__ W,
                     _Float16* __restrict__ dst, float scale)
{
  __shared__ _Float16 xl[128][40];  // [row][k], 32 + 8 pad
  __shared__ _Float16 wl[64][40];   // [n][k]  (W^T tile), 32 + 8 pad

  const int tid  = threadIdx.x;
  const int wave = tid >> 5;
  const int lane = tid & 31;
  const int l16  = lane & 15;
  const int hi   = (lane >> 4) & 1;
  const long rowBase = (long)blockIdx.x * 128;

  v8f acc[4];
#pragma unroll
  for (int nt = 0; nt < 4; ++nt) acc[nt] = (v8f){0, 0, 0, 0, 0, 0, 0, 0};

  for (int kt = 0; kt < Cc; kt += 32) {
    // stage x tile: 128x32, 16 elements/thread, fp32 -> f16
    {
      const int r  = tid >> 1;
      const int kh = (tid & 1) * 16;
      const float* src = x + (rowBase + r) * Cc + kt + kh;
#pragma unroll
      for (int j = 0; j < 16; ++j) xl[r][kh + j] = (_Float16)src[j];
      if (kt + 32 < Cc) __builtin_prefetch(src + 32, 0, 3);  // next k-panel
    }
    // stage W^T tile: 64x32, 8 elements/thread (contiguous along c)
    {
      const int n  = tid & 63;
      const int ko = (tid >> 6) * 8;
      const float* src = W + n * Cc + kt + ko;
#pragma unroll
      for (int j = 0; j < 8; ++j) wl[n][ko + j] = (_Float16)src[j];
    }
    __syncthreads();

    // A fragment (16x32): row = lane&15; lanes 0-15 hold K 0-15, 16-31 K 16-31
    v16h afrag;
    {
      const _Float16* ap = &xl[wave * 16 + l16][hi * 16];
#pragma unroll
      for (int h = 0; h < 8; ++h) afrag[h] = ap[h];
#pragma unroll
      for (int h = 0; h < 8; ++h) afrag[8 + h] = ap[8 + h];
    }
#pragma unroll
    for (int nt = 0; nt < 4; ++nt) {
      // B fragment (32x16): col = lane&15; lanes 0-15 K 0-15, lanes 16-31 K 16-31
      v16h bfrag;
      const _Float16* bp = &wl[nt * 16 + l16][hi * 16];
#pragma unroll
      for (int h = 0; h < 16; ++h) bfrag[h] = bp[h];
      acc[nt] = WMMA_F32_F16(afrag, bfrag, acc[nt]);
    }
    __syncthreads();
  }

  // C/D layout: VGPR r -> row r + 8*hi, col = lane&15
#pragma unroll
  for (int nt = 0; nt < 4; ++nt) {
#pragma unroll
    for (int r = 0; r < 8; ++r) {
      const long row = rowBase + wave * 16 + r + 8 * hi;
      dst[row * Dd + nt * 16 + l16] = (_Float16)(acc[nt][r] * scale);
    }
  }
}

// ---------------------------------------------------------------------------
// Kernel 2: causal flash attention, S^T formulation.
// block = 256 threads (8 waves), 128 query rows per block (16 per wave).
// S^T = K (keys x d) @ Q^T (d x queries): keys on M, queries on N, so each
// lane owns ONE query column -> softmax is per-lane + one partner shuffle,
// and the P (f32 C-layout) -> A-fragment relayout is a pure register permute.
// K/V key-blocks of 64 double-buffered in LDS via async global->LDS copies.
// ---------------------------------------------------------------------------
__global__ __launch_bounds__(256)
void flash_attn_kernel(const _Float16* __restrict__ Qf,
                       const _Float16* __restrict__ Kf,
                       const _Float16* __restrict__ Vf,
                       float* __restrict__ out)
{
  // stride 72 f16 = 144 B: 16B-aligned rows (needed by b128 async copies)
  __shared__ _Float16 Kl[2][64][72];  // [buf][key][d]
  __shared__ _Float16 Vl[2][64][72];  // [buf][key][d]

  const int tid  = threadIdx.x;
  const int wave = tid >> 5;
  const int lane = tid & 31;
  const int l16  = lane & 15;
  const int hi   = (lane >> 4) & 1;
  const int b     = blockIdx.y;
  const int qBase = blockIdx.x * 128;
  const int qrow0 = qBase + wave * 16;
  const long bOff = (long)b * Tt * Dd;

  // per-thread staging slice: 32 B of K and 32 B of V per tile
  const int skey  = tid >> 2;
  const int sdoff = (tid & 3) * 16;
  const _Float16* gK0 = Kf + bOff + (long)skey * Dd + sdoff;
  const _Float16* gV0 = Vf + bOff + (long)skey * Dd + sdoff;

  // Resident Q^T B-fragments: col = query (lane&15); lanes lo K 0-15, hi 16-31.
  // Q already carries 1/sqrt(D).
  v16h qb[2];
  {
    const _Float16* qp = Qf + bOff + (long)(qrow0 + l16) * Dd + hi * 16;
#pragma unroll
    for (int dc = 0; dc < 2; ++dc)
#pragma unroll
      for (int h = 0; h < 16; ++h) qb[dc][h] = qp[dc * 32 + h];
  }

  v8f   o[4];
  float m = -1e30f, lsum = 0.0f;  // per-lane: this lane's query column
#pragma unroll
  for (int nt = 0; nt < 4; ++nt) o[nt] = (v8f){0, 0, 0, 0, 0, 0, 0, 0};

  const int nkb = qBase / 64 + 2;  // causal: last key block covers qBase+64..+127

  // prime the pipeline: tile 0 -> buffer 0 (4 async b128 copies per thread)
  async_cp32B(gK0, &Kl[0][skey][sdoff]);
  async_cp32B(gV0, &Vl[0][skey][sdoff]);

  for (int kb = 0; kb < nkb; ++kb) {
    const int cur = kb & 1;
    const bool more = (kb + 1) < nkb;
    if (more) {  // stream next tile into the other buffer (freed by last barrier)
      const long off = (long)(kb + 1) * 64 * Dd;
      async_cp32B(gK0 + off, &Kl[1 - cur][skey][sdoff]);
      async_cp32B(gV0 + off, &Vl[1 - cur][skey][sdoff]);
    }
    // wait for *current* tile: <= 4 outstanding means only next-tile copies left
    if (more) { WAIT_ASYNC(4); } else { WAIT_ASYNC(0); }
    __syncthreads();  // every wave has waited -> both tiles fully in LDS

    const int key0 = kb * 64;
    if (key0 <= qrow0 + 15) {  // wave has at least one unmasked element
      // ---- S^T = K Q^T : 4 key tiles (M) x 16 queries (N) ----
      v8f st[4];
#pragma unroll
      for (int kt = 0; kt < 4; ++kt) {
        st[kt] = (v8f){0, 0, 0, 0, 0, 0, 0, 0};
        const _Float16* ar = &Kl[cur][kt * 16 + l16][0];  // A row = key
#pragma unroll
        for (int dc = 0; dc < 2; ++dc) {
          v16h ka;  // A halves: lanes lo d 0-15, hi 16-31 (+dc*32)
#pragma unroll
          for (int h = 0; h < 8; ++h) ka[h]     = ar[dc * 32 + hi * 16 + h];
#pragma unroll
          for (int h = 0; h < 8; ++h) ka[8 + h] = ar[dc * 32 + hi * 16 + 8 + h];
          st[kt] = WMMA_F32_F16(ka, qb[dc], st[kt]);
        }
      }
      // ---- causal mask (diagonal region only) ----
      if (key0 + 63 > qrow0) {
        const int q = qrow0 + l16;
#pragma unroll
        for (int kt = 0; kt < 4; ++kt)
#pragma unroll
          for (int r = 0; r < 8; ++r) {
            const int key = key0 + kt * 16 + r + 8 * hi;
            if (key > q) st[kt][r] = -1e30f;
          }
      }
      // ---- online softmax: lane = one query; partner lane (xor 16) has the
      //      other half of the keys ----
      float vmax = st[0][0];
#pragma unroll
      for (int kt = 0; kt < 4; ++kt)
#pragma unroll
        for (int r = 0; r < 8; ++r) vmax = fmaxf(vmax, st[kt][r]);
      vmax = fmaxf(vmax, __shfl_xor(vmax, 16, 32));
      const float mnew = fmaxf(m, vmax);
      const float corr = __expf(m - mnew);
      m = mnew;

      float rsum = 0.0f;
#pragma unroll
      for (int kt = 0; kt < 4; ++kt)
#pragma unroll
        for (int r = 0; r < 8; ++r) {
          const float p = __expf(st[kt][r] - mnew);
          st[kt][r] = p;
          rsum += p;
        }
      rsum += __shfl_xor(rsum, 16, 32);
      lsum = lsum * corr + rsum;

      // broadcast corr to the lanes owning each O row (O row = query r+8*hi)
      float corrR[8];
#pragma unroll
      for (int r = 0; r < 8; ++r) corrR[r] = __shfl(corr, r + 8 * hi, 32);
#pragma unroll
      for (int nt = 0; nt < 4; ++nt)
#pragma unroll
        for (int r = 0; r < 8; ++r) o[nt][r] *= corrR[r];

      // ---- P A-fragments: pure register permutation + f32->f16 convert ----
      // chunk kc keys kc*32..+31: lanes lo need keys {0-7,16-23} = st[2kc][h],
      // st[2kc+1][h]; lanes hi need {8-15,24-31} = same regs (they hold r+8).
      v16h pa[2];
#pragma unroll
      for (int kc = 0; kc < 2; ++kc) {
#pragma unroll
        for (int h = 0; h < 8; ++h) pa[kc][h]     = (_Float16)st[2 * kc][h];
#pragma unroll
        for (int h = 0; h < 8; ++h) pa[kc][8 + h] = (_Float16)st[2 * kc + 1][h];
      }
      // ---- O += P @ V ----
#pragma unroll
      for (int nt = 0; nt < 4; ++nt) {    // output d tile
#pragma unroll
        for (int kc = 0; kc < 2; ++kc) {  // key chunk of 32
          v16h vb;  // B = V: col = d (lane&15), K = key -> strided b16 gather
          const _Float16* bp = &Vl[cur][kc * 32 + hi * 8][nt * 16 + l16];
#pragma unroll
          for (int h = 0; h < 16; ++h) vb[h] = bp[(h + (h & 8)) * 72];
          o[nt] = WMMA_F32_F16(pa[kc], vb, o[nt]);
        }
      }
    }
    __syncthreads();  // everyone done reading buf[cur] before it is refilled
  }

  // ---- normalize and store fp32 (O row = query r + 8*hi) ----
  float lR[8];
#pragma unroll
  for (int r = 0; r < 8; ++r) lR[r] = __shfl(lsum, r + 8 * hi, 32);
#pragma unroll
  for (int nt = 0; nt < 4; ++nt)
#pragma unroll
    for (int r = 0; r < 8; ++r) {
      const long row = qrow0 + r + 8 * hi;
      out[bOff + row * Dd + nt * 16 + l16] = o[nt][r] / lR[r];
    }
}

// ---------------------------------------------------------------------------
extern "C" void kernel_launch(void* const* d_in, const int* in_sizes, int n_in,
                              void* d_out, int out_size, void* d_ws, size_t ws_size,
                              hipStream_t stream)
{
  const float* x  = (const float*)d_in[0];
  const float* Wk = (const float*)d_in[1];
  const float* Wq = (const float*)d_in[2];
  const float* Wv = (const float*)d_in[3];
  float* out = (float*)d_out;

  // workspace: Q,K,V in f16, B*T*D elements each (2 MB each, 6 MB total)
  _Float16* Qh = (_Float16*)d_ws;
  _Float16* Kh = Qh + (size_t)Bb * Tt * Dd;
  _Float16* Vh = Kh + (size_t)Bb * Tt * Dd;

  dim3 blk(256, 1, 1);
  dim3 gproj((Bb * Tt) / 128, 1, 1);
  qkv_proj_kernel<<<gproj, blk, 0, stream>>>(x, Wq, Qh, 0.125f);  // fold 1/sqrt(D)
  qkv_proj_kernel<<<gproj, blk, 0, stream>>>(x, Wk, Kh, 1.0f);
  qkv_proj_kernel<<<gproj, blk, 0, stream>>>(x, Wv, Vh, 1.0f);

  dim3 gattn(Tt / 128, Bb, 1);
  flash_attn_kernel<<<gattn, blk, 0, stream>>>(Qh, Kh, Vh, out);
}
